// GAT14_20693152432425
// MI455X (gfx1250) — compile-verified, matrix operand order checked by reading, not run on hardware
//
#include <hip/hip_runtime.h>
#include <cmath>

// ---------------------------------------------------------------------------
// Problem constants (match reference)
// ---------------------------------------------------------------------------
#define NN   10000     // nodes
#define FIN  128       // input features
#define CC   256       // channels per head
#define HH   4         // heads (layers 1,2)
#define EE   160000    // edges
#define GG   512       // graphs
#define HC   1024      // HH*CC
#define BN_EPS 1e-5f

// ---------------------------------------------------------------------------
// WMMA types (gfx1250, wave32)
// ---------------------------------------------------------------------------
typedef __bf16 v16bf __attribute__((ext_vector_type(16)));
typedef float  v8f   __attribute__((ext_vector_type(8)));

// ---------------------------------------------------------------------------
// Helpers
// ---------------------------------------------------------------------------
__device__ inline unsigned short f2bf(float f) {
    unsigned int u = __float_as_uint(f);
    return (unsigned short)((u + 0x7FFFu + ((u >> 16) & 1u)) >> 16);   // RNE
}

__device__ inline void atomicMaxF(float* addr, float val) {
    int* ai = (int*)addr;
    int old = __float_as_int(*addr);
    while (__int_as_float(old) < val) {
        int prev = atomicCAS(ai, old, __float_as_int(val));
        if (prev == old) break;
        old = prev;
    }
}

// ---------------------------------------------------------------------------
// Elementwise conversion kernels
// ---------------------------------------------------------------------------
__global__ void k_f32_to_bf16(const float* __restrict__ in,
                              unsigned short* __restrict__ out, size_t n) {
    size_t i = (size_t)blockIdx.x * blockDim.x + threadIdx.x;
    if (i < n) out[i] = f2bf(in[i]);
}

// W [K, Nc] fp32  ->  Wt [Nc, K] bf16
__global__ void k_transpose_bf16(const float* __restrict__ W,
                                 unsigned short* __restrict__ Wt, int K, int Nc) {
    int i = blockIdx.x * blockDim.x + threadIdx.x;
    if (i >= K * Nc) return;
    int k = i / Nc, nc = i % Nc;
    Wt[(size_t)nc * K + k] = f2bf(W[i]);
}

__global__ void k_fill_f32(float* __restrict__ p, float v, size_t n) {
    size_t i = (size_t)blockIdx.x * blockDim.x + threadIdx.x;
    if (i < n) p[i] = v;
}

// ---------------------------------------------------------------------------
// bf16 WMMA GEMM:  C[M,Nc] = A[M,K] (bf16) x Bt[Nc,K]^T (bf16),  fp32 out.
// One wave computes a 16x64 output strip (4 accumulators) so the A fragment
// is loaded once per K-step and reused 4x: 10 b128 loads : 4 WMMA.
// Requires M%16==0, K%32==0, Nc%64==0.
// Fragment layout (16-bit A 16x32, ISA 7.12.2): lane L -> row/col L%16,
// half = L/16; elements 0..7 = K[k0 + 8*half .. +7], 8..15 = K[k0+16+8*half..+7]
// -> two contiguous 16B loads per fragment.
// ---------------------------------------------------------------------------
__global__ void k_wmma_gemm(const unsigned short* __restrict__ A,
                            const unsigned short* __restrict__ Bt,
                            float* __restrict__ C,
                            int M, int K, int Nc) {
    const int lane = threadIdx.x & 31;
    const int wid  = blockIdx.x * (blockDim.x >> 5) + (threadIdx.x >> 5);
    const int Nt64 = Nc >> 6;                   // 64-wide N strips
    const int Mt   = M >> 4;
    const int tm   = wid / Nt64;
    const int tn   = wid % Nt64;
    if (tm >= Mt) return;                       // wave-uniform guard

    const int r  = lane & 15;
    const int hi = lane >> 4;
    const unsigned short* arow = A  + (size_t)(tm * 16 + r) * K + hi * 8;
    const unsigned short* brow = Bt + (size_t)(tn * 64 + r) * K + hi * 8;
    const size_t bstride = (size_t)16 * K;      // next 16-col sub-tile of B

    v8f acc0 = {}, acc1 = {}, acc2 = {}, acc3 = {};
    for (int k0 = 0; k0 < K; k0 += 32) {
        // prefetch next K-block (speculative; OOB prefetch is dropped)
        __builtin_prefetch(arow + k0 + 32, 0, 0);
        __builtin_prefetch(brow + k0 + 32, 0, 0);

        union { v16bf v; uint4 q[2]; } fa, fb0, fb1, fb2, fb3;
        fa.q[0]  = *(const uint4*)(arow + k0);
        fa.q[1]  = *(const uint4*)(arow + k0 + 16);
        fb0.q[0] = *(const uint4*)(brow + k0);
        fb0.q[1] = *(const uint4*)(brow + k0 + 16);
        fb1.q[0] = *(const uint4*)(brow + bstride + k0);
        fb1.q[1] = *(const uint4*)(brow + bstride + k0 + 16);
        fb2.q[0] = *(const uint4*)(brow + 2 * bstride + k0);
        fb2.q[1] = *(const uint4*)(brow + 2 * bstride + k0 + 16);
        fb3.q[0] = *(const uint4*)(brow + 3 * bstride + k0);
        fb3.q[1] = *(const uint4*)(brow + 3 * bstride + k0 + 16);

        acc0 = __builtin_amdgcn_wmma_f32_16x16x32_bf16(
                   false, fa.v, false, fb0.v, (short)0, acc0, false, false);
        acc1 = __builtin_amdgcn_wmma_f32_16x16x32_bf16(
                   false, fa.v, false, fb1.v, (short)0, acc1, false, false);
        acc2 = __builtin_amdgcn_wmma_f32_16x16x32_bf16(
                   false, fa.v, false, fb2.v, (short)0, acc2, false, false);
        acc3 = __builtin_amdgcn_wmma_f32_16x16x32_bf16(
                   false, fa.v, false, fb3.v, (short)0, acc3, false, false);
    }

    // C/D layout: VGPR i -> row tm*16 + hi*8 + i, col = strip base + r
    float* crow = C + (size_t)(tm * 16 + hi * 8) * Nc + tn * 64 + r;
#pragma unroll
    for (int i = 0; i < 8; ++i) {
        float* p = crow + (size_t)i * Nc;
        p[0]  = acc0[i];
        p[16] = acc1[i];
        p[32] = acc2[i];
        p[48] = acc3[i];
    }
}

// ---------------------------------------------------------------------------
// a_src[n,h] = <h[n,h,:], att_src[h,:]> ; a_dst likewise. One wave per (n,h).
// ---------------------------------------------------------------------------
__global__ void k_att_dots(const float* __restrict__ hfeat,
                           const float* __restrict__ att_src,
                           const float* __restrict__ att_dst,
                           float* __restrict__ asrc, float* __restrict__ adst,
                           int n_nodes, int heads, int chan) {
    int lane = threadIdx.x & 31;
    int wid  = blockIdx.x * (blockDim.x >> 5) + (threadIdx.x >> 5);
    if (wid >= n_nodes * heads) return;
    int h = wid % heads;
    const float* hp = hfeat + (size_t)wid * chan;        // [n,h] row
    const float* as = att_src + (size_t)h * chan;
    const float* ad = att_dst + (size_t)h * chan;
    float s = 0.f, d = 0.f;
    for (int c = lane; c < chan; c += 32) { float v = hp[c]; s += v * as[c]; d += v * ad[c]; }
#pragma unroll
    for (int o = 16; o; o >>= 1) { s += __shfl_down(s, o, 32); d += __shfl_down(d, o, 32); }
    if (lane == 0) { asrc[wid] = s; adst[wid] = d; }
}

// ---------------------------------------------------------------------------
// Edge pass 1: leaky-relu logits + scatter max into m[dst,h]
// ---------------------------------------------------------------------------
__global__ void k_edge_logits(const int* __restrict__ ei,
                              const float* __restrict__ asrc,
                              const float* __restrict__ adst,
                              float* __restrict__ logit, float* __restrict__ m,
                              int nE, int heads) {
    int i = blockIdx.x * blockDim.x + threadIdx.x;
    if (i >= nE * heads) return;
    int e = i / heads, h = i % heads;
    int s = ei[e], d = ei[nE + e];
    float l = asrc[s * heads + h] + adst[d * heads + h];
    l = l > 0.f ? l : 0.2f * l;
    logit[i] = l;
    atomicMaxF(&m[d * heads + h], l);
}

__global__ void k_fix_m(float* __restrict__ m, int n) {
    int i = blockIdx.x * blockDim.x + threadIdx.x;
    if (i < n && isinf(m[i])) m[i] = 0.f;
}

// Edge pass 2: e = exp(logit - m[dst]); scatter-sum denom
__global__ void k_edge_exp(const int* __restrict__ ei,
                           float* __restrict__ logit, const float* __restrict__ m,
                           float* __restrict__ denom, int nE, int heads) {
    int i = blockIdx.x * blockDim.x + threadIdx.x;
    if (i >= nE * heads) return;
    int e = i / heads, h = i % heads;
    int d = ei[nE + e];
    float ev = expf(logit[i] - m[d * heads + h]);
    logit[i] = ev;
    atomicAdd(&denom[d * heads + h], ev);
}

// Edge pass 3: out[dst,h,:] += alpha * h[src,h,:].  One wave per (edge,head).
__global__ void k_edge_aggregate(const int* __restrict__ ei,
                                 const float* __restrict__ hfeat,
                                 const float* __restrict__ evals,
                                 const float* __restrict__ denom,
                                 float* __restrict__ aggr,
                                 int nE, int heads, int chan) {
    int lane = threadIdx.x & 31;
    int wid  = blockIdx.x * (blockDim.x >> 5) + (threadIdx.x >> 5);
    if (wid >= nE * heads) return;
    int e = wid / heads, h = wid % heads;
    int s = ei[e], d = ei[nE + e];
    float alpha = evals[wid] / (denom[d * heads + h] + 1e-16f);
    const float* hp = hfeat + ((size_t)s * heads + h) * chan;
    float*       op = aggr  + ((size_t)d * heads + h) * chan;
    for (int c = lane; c < chan; c += 32) atomicAdd(&op[c], alpha * hp[c]);
}

// ---------------------------------------------------------------------------
// Fused bias + eval-mode BatchNorm + activation. mode: 0=ELU, 1=ReLU, 2=none
// bn is [4, cols]: gamma, beta, mean, var
// ---------------------------------------------------------------------------
__global__ void k_bias_bn_act(const float* __restrict__ in,
                              const float* __restrict__ bias,
                              const float* __restrict__ bn,
                              float* __restrict__ out,
                              int rows, int cols, int mode) {
    size_t i = (size_t)blockIdx.x * blockDim.x + threadIdx.x;
    if (i >= (size_t)rows * cols) return;
    int j = (int)(i % cols);
    float v = in[i] + (bias ? bias[j] : 0.f);
    float g = bn[j], b = bn[cols + j], mu = bn[2 * cols + j], var = bn[3 * cols + j];
    v = (v - mu) * (g * rsqrtf(var + BN_EPS)) + b;
    if (mode == 0)      v = v > 0.f ? v : (expf(v) - 1.f);
    else if (mode == 1) v = v > 0.f ? v : 0.f;
    out[i] = v;
}

// ---------------------------------------------------------------------------
// Readout pooling
// ---------------------------------------------------------------------------
__global__ void k_pool_scatter(const float* __restrict__ h, const int* __restrict__ batch,
                               float* __restrict__ s, float* __restrict__ mx,
                               float* __restrict__ cnt, int n, int chan) {
    int i = blockIdx.x * blockDim.x + threadIdx.x;
    if (i >= n * chan) return;
    int node = i / chan, c = i % chan;
    int g = batch[node];
    float v = h[i];
    atomicAdd(&s[(size_t)g * chan + c], v);
    atomicMaxF(&mx[(size_t)g * chan + c], v);
    if (c == 0) atomicAdd(&cnt[g], 1.f);
}

__global__ void k_pool_concat(const float* __restrict__ s, const float* __restrict__ mx,
                              const float* __restrict__ cnt, float* __restrict__ z,
                              int ngraph, int chan) {
    int i = blockIdx.x * blockDim.x + threadIdx.x;
    if (i >= ngraph * chan) return;
    int g = i / chan, c = i % chan;
    float ss = s[i];
    z[(size_t)g * 3 * chan + c]            = ss / cnt[g];
    z[(size_t)g * 3 * chan + chan + c]     = mx[i];
    z[(size_t)g * 3 * chan + 2 * chan + c] = ss;
}

// Final [G,chan] @ [chan,1] + b : one wave per graph
__global__ void k_final_fc(const float* __restrict__ z, const float* __restrict__ w,
                           const float* __restrict__ b, float* __restrict__ out,
                           int ngraph, int chan) {
    int lane = threadIdx.x & 31;
    int g = blockIdx.x * (blockDim.x >> 5) + (threadIdx.x >> 5);
    if (g >= ngraph) return;
    float s = 0.f;
    for (int c = lane; c < chan; c += 32) s += z[(size_t)g * chan + c] * w[c];
#pragma unroll
    for (int o = 16; o; o >>= 1) s += __shfl_down(s, o, 32);
    if (lane == 0) out[g] = s + b[0];
}

// ---------------------------------------------------------------------------
// Host orchestration
// ---------------------------------------------------------------------------
extern "C" void kernel_launch(void* const* d_in, const int* in_sizes, int n_in,
                              void* d_out, int out_size, void* d_ws, size_t ws_size,
                              hipStream_t stream) {
    (void)in_sizes; (void)n_in; (void)out_size; (void)ws_size;

    // ---- inputs (setup_inputs dict order) ----
    const float* x        = (const float*)d_in[0];
    const int*   ei       = (const int*)d_in[1];     // [2,E]
    const int*   batch    = (const int*)d_in[2];     // [N]
    const float* W1       = (const float*)d_in[3];
    const float* att_s1   = (const float*)d_in[4];
    const float* att_d1   = (const float*)d_in[5];
    const float* b1       = (const float*)d_in[6];
    const float* bn1      = (const float*)d_in[7];
    const float* W2       = (const float*)d_in[8];
    const float* att_s2   = (const float*)d_in[9];
    const float* att_d2   = (const float*)d_in[10];
    const float* b2       = (const float*)d_in[11];
    const float* bn2      = (const float*)d_in[12];
    const float* W3       = (const float*)d_in[13];
    const float* att_s3   = (const float*)d_in[14];
    const float* att_d3   = (const float*)d_in[15];
    const float* b3       = (const float*)d_in[16];
    const float* bn3      = (const float*)d_in[17];
    const float* fc1_w    = (const float*)d_in[18];
    const float* fc1_b    = (const float*)d_in[19];
    const float* bnf1     = (const float*)d_in[20];
    const float* fc2_w    = (const float*)d_in[21];
    const float* fc2_b    = (const float*)d_in[22];
    const float* bnf2     = (const float*)d_in[23];
    const float* fc3_w    = (const float*)d_in[24];
    const float* fc3_b    = (const float*)d_in[25];
    float* out = (float*)d_out;

    // ---- workspace carve ----
    char* ws = (char*)d_ws;
    size_t off = 0;
    auto carve = [&](size_t bytes) -> void* {
        void* p = ws + off;
        off += (bytes + 255) & ~(size_t)255;
        return p;
    };
    float*          f_hfeat = (float*)carve((size_t)NN * HC * 4);      // pre-aggregation features
    float*          f_aggr  = (float*)carve((size_t)NN * HC * 4);      // scatter-sum target
    float*          f_act   = (float*)carve((size_t)NN * HC * 4);      // layer activation output
    unsigned short* bf_act  = (unsigned short*)carve((size_t)NN * HC * 2);
    unsigned short* bf_wt   = (unsigned short*)carve((size_t)HC * HC * 2);
    float*          f_asrc  = (float*)carve((size_t)NN * HH * 4);
    float*          f_adst  = (float*)carve((size_t)NN * HH * 4);
    float*          f_logit = (float*)carve((size_t)EE * HH * 4);      // logits, then exp-values
    float*          f_m     = (float*)carve((size_t)NN * HH * 4);
    float*          f_denom = (float*)carve((size_t)NN * HH * 4);
    float*          f_s     = (float*)carve((size_t)GG * CC * 4);
    float*          f_mx    = (float*)carve((size_t)GG * CC * 4);
    float*          f_cnt   = (float*)carve((size_t)GG * 4);
    float*          f_z     = (float*)carve((size_t)GG * 3 * CC * 4);
    unsigned short* bf_z    = (unsigned short*)carve((size_t)GG * 3 * CC * 2);
    float*          f_z2    = (float*)carve((size_t)GG * 2 * CC * 4);
    float*          f_z3    = (float*)carve((size_t)GG * CC * 4);

    const float NEG_INF = -INFINITY;
    auto nb = [](size_t n, int bs) { return (unsigned)((n + bs - 1) / bs); };

    // ---- one GAT layer ----
    auto run_gat = [&](const float* act_in, int Kin,
                       const float* W, const float* att_s, const float* att_d,
                       const float* bias, const float* bn, int heads, float* act_out) {
        const int outc = heads * CC;
        k_f32_to_bf16<<<nb((size_t)NN * Kin, 256), 256, 0, stream>>>(act_in, bf_act, (size_t)NN * Kin);
        k_transpose_bf16<<<nb((size_t)Kin * outc, 256), 256, 0, stream>>>(W, bf_wt, Kin, outc);
        const int tiles = (NN / 16) * (outc / 64);
        k_wmma_gemm<<<nb(tiles, 8), 256, 0, stream>>>(bf_act, bf_wt, f_hfeat, NN, Kin, outc);
        k_att_dots<<<nb((size_t)NN * heads, 8), 256, 0, stream>>>(f_hfeat, att_s, att_d,
                                                                  f_asrc, f_adst, NN, heads, CC);
        k_fill_f32<<<nb((size_t)NN * heads, 256), 256, 0, stream>>>(f_m, NEG_INF, (size_t)NN * heads);
        k_fill_f32<<<nb((size_t)NN * heads, 256), 256, 0, stream>>>(f_denom, 0.f, (size_t)NN * heads);
        k_fill_f32<<<nb((size_t)NN * outc, 256), 256, 0, stream>>>(f_aggr, 0.f, (size_t)NN * outc);
        k_edge_logits<<<nb((size_t)EE * heads, 256), 256, 0, stream>>>(ei, f_asrc, f_adst,
                                                                       f_logit, f_m, EE, heads);
        k_fix_m<<<nb((size_t)NN * heads, 256), 256, 0, stream>>>(f_m, NN * heads);
        k_edge_exp<<<nb((size_t)EE * heads, 256), 256, 0, stream>>>(ei, f_logit, f_m, f_denom, EE, heads);
        k_edge_aggregate<<<nb((size_t)EE * heads, 8), 256, 0, stream>>>(ei, f_hfeat, f_logit,
                                                                        f_denom, f_aggr, EE, heads, CC);
        k_bias_bn_act<<<nb((size_t)NN * outc, 256), 256, 0, stream>>>(f_aggr, bias, bn, act_out,
                                                                      NN, outc, /*ELU*/0);
    };

    // ---- three GAT layers ----
    run_gat(x,     FIN, W1, att_s1, att_d1, b1, bn1, HH, f_act);   // [N,1024]
    run_gat(f_act, HC,  W2, att_s2, att_d2, b2, bn2, HH, f_act);   // [N,1024]
    run_gat(f_act, HC,  W3, att_s3, att_d3, b3, bn3, 1,  f_act);   // [N,256]

    // ---- readout pooling: mean | max | sum -> z [G, 768] ----
    k_fill_f32<<<nb((size_t)GG * CC, 256), 256, 0, stream>>>(f_s, 0.f, (size_t)GG * CC);
    k_fill_f32<<<nb((size_t)GG * CC, 256), 256, 0, stream>>>(f_mx, NEG_INF, (size_t)GG * CC);
    k_fill_f32<<<nb((size_t)GG, 256), 256, 0, stream>>>(f_cnt, 0.f, (size_t)GG);
    k_pool_scatter<<<nb((size_t)NN * CC, 256), 256, 0, stream>>>(f_act, batch, f_s, f_mx, f_cnt, NN, CC);
    k_pool_concat<<<nb((size_t)GG * CC, 256), 256, 0, stream>>>(f_s, f_mx, f_cnt, f_z, GG, CC);

    // ---- FC1: [512,768] @ [768,512] + BN + ReLU ----
    k_f32_to_bf16<<<nb((size_t)GG * 3 * CC, 256), 256, 0, stream>>>(f_z, bf_z, (size_t)GG * 3 * CC);
    k_transpose_bf16<<<nb((size_t)3 * CC * 2 * CC, 256), 256, 0, stream>>>(fc1_w, bf_wt, 3 * CC, 2 * CC);
    k_wmma_gemm<<<nb((GG / 16) * (2 * CC / 64), 8), 256, 0, stream>>>(bf_z, bf_wt, f_hfeat, GG, 3 * CC, 2 * CC);
    k_bias_bn_act<<<nb((size_t)GG * 2 * CC, 256), 256, 0, stream>>>(f_hfeat, fc1_b, bnf1, f_z2,
                                                                    GG, 2 * CC, /*ReLU*/1);

    // ---- FC2: [512,512] @ [512,256] + BN + ReLU ----
    k_f32_to_bf16<<<nb((size_t)GG * 2 * CC, 256), 256, 0, stream>>>(f_z2, bf_z, (size_t)GG * 2 * CC);
    k_transpose_bf16<<<nb((size_t)2 * CC * CC, 256), 256, 0, stream>>>(fc2_w, bf_wt, 2 * CC, CC);
    k_wmma_gemm<<<nb((GG / 16) * (CC / 64), 8), 256, 0, stream>>>(bf_z, bf_wt, f_hfeat, GG, 2 * CC, CC);
    k_bias_bn_act<<<nb((size_t)GG * CC, 256), 256, 0, stream>>>(f_hfeat, fc2_b, bnf2, f_z3,
                                                                GG, CC, /*ReLU*/1);

    // ---- FC3: [512,256] @ [256,1] + b -> out [512] ----
    k_final_fc<<<nb((size_t)GG, 8), 256, 0, stream>>>(f_z3, fc3_w, fc3_b, out, GG, CC);
}
